// AttentionOpWithKVCache_31507880083461
// MI455X (gfx1250) — matrix operand order repaired and checked by reading, not compiled, not run on hardware
//
#include <hip/hip_runtime.h>

typedef __attribute__((ext_vector_type(16))) __bf16         v16bf;
typedef __attribute__((ext_vector_type(8)))  float          v8f;
typedef __attribute__((ext_vector_type(16))) unsigned short v16u;
typedef __attribute__((ext_vector_type(8)))  unsigned short v8u;

#define B_    2
#define T_    1024
#define H_    16
#define D_    128
#define F_    16
#define MAXC_ 4

#define LDQ 136   // 128 + 8 pad (bf16 elems)
#define LDK 136
#define LDV 40    // 32 + 8 pad
#define LDP 40
#define KT  32    // keys per tile
#define TPC 32    // tiles per chunk = T_/KT

__device__ __forceinline__ unsigned short f2bf(float x) {
  unsigned int u = __float_as_uint(x);
  u += 0x7FFFu + ((u >> 16) & 1u);   // RNE
  return (unsigned short)(u >> 16);
}

#if __has_builtin(__builtin_amdgcn_cvt_pk_bf16_f32)
__device__ __forceinline__ unsigned int pk2bf(float x, float y) {
  auto t = __builtin_amdgcn_cvt_pk_bf16_f32(x, y);   // single v_cvt_pk_bf16_f32
  return __builtin_bit_cast(unsigned int, t);
}
#else
__device__ __forceinline__ unsigned int pk2bf(float x, float y) {
  return (unsigned int)f2bf(x) | ((unsigned int)f2bf(y) << 16);
}
#endif

#if __has_builtin(__builtin_amdgcn_sched_barrier)
#define SCHED_FENCE() __builtin_amdgcn_sched_barrier(0)
#else
#define SCHED_FENCE()
#endif

// 16-lane (half-wave) max reduction on the VALU pipe via DPP row_ror rotations
#if __has_builtin(__builtin_amdgcn_mov_dpp)
template <int CTRL>
__device__ __forceinline__ float dpp_ror(float x) {
  return __builtin_bit_cast(float,
      __builtin_amdgcn_mov_dpp(__builtin_bit_cast(int, x), CTRL, 0xF, 0xF, false));
}
__device__ __forceinline__ float red_max16(float v) {
  v = fmaxf(v, dpp_ror<0x121>(v));  // row_ror:1
  v = fmaxf(v, dpp_ror<0x122>(v));  // row_ror:2
  v = fmaxf(v, dpp_ror<0x124>(v));  // row_ror:4
  v = fmaxf(v, dpp_ror<0x128>(v));  // row_ror:8
  return v;
}
#else
__device__ __forceinline__ float red_max16(float v) {
  for (int off = 1; off < 16; off <<= 1) v = fmaxf(v, __shfl_xor(v, off, 32));
  return v;
}
#endif

__device__ __forceinline__ v16u cat8(v8u lo, v8u hi) {
  return __builtin_shufflevector(lo, hi, 0,1,2,3,4,5,6,7,8,9,10,11,12,13,14,15);
}

// ---------------- cache update + pass-through copy (pure bandwidth) ----------------
__global__ __launch_bounds__(256)
void cache_copy_kernel(const float4* __restrict__ kc, const float4* __restrict__ vc,
                       const float4* __restrict__ k,  const float4* __restrict__ v,
                       const int* __restrict__ cur_idx, const int* __restrict__ store_kv,
                       float4* __restrict__ out_kc, float4* __restrict__ out_vc) {
  const long frame = (long)B_ * T_ * H_ * D_ / 4;
  long i = (long)blockIdx.x * blockDim.x + threadIdx.x;
  const int idx = *cur_idx, sk = *store_kv;
  const int f = (int)(i / frame);
  const long r = i - (long)f * frame;
  const bool repl = (sk != 0) && (f == idx);
  if (blockIdx.y == 0) out_kc[i] = repl ? k[r] : kc[i];
  else                 out_vc[i] = repl ? v[r] : vc[i];
}

// ---------------- flash attention over (cached frames + current) ----------------
__global__ __launch_bounds__(128)
void flash_attn_kernel(const float* __restrict__ q,
                       const float* __restrict__ kin, const float* __restrict__ vin,
                       const float* __restrict__ kc,  const float* __restrict__ vc,
                       const int* __restrict__ cur_idx, const int* __restrict__ run_kv,
                       float* __restrict__ out) {
  __shared__ __align__(16) unsigned short sQ[64 * LDQ];
  __shared__ __align__(16) unsigned short sK[2][KT * LDK];
  __shared__ __align__(16) unsigned short sVt[2][D_ * LDV];  // transposed: [d][key]
  __shared__ __align__(16) unsigned short sP[4 * 16 * LDP];

  const int tid  = threadIdx.x;
  const int wave = tid >> 5;
  const int lane = tid & 31;
  const int mrow = lane & 15;        // row/col within 16-lane half
  const int hi16 = (lane >> 4) & 1;  // which half of the wave
  const int abase = hi16 * 8;        // A-matrix K-chunk base (ISA 16-bit A layout)

  const int q0 = blockIdx.x * 64;
  const int h  = blockIdx.y;
  const int b  = blockIdx.z;
  const long bstride = (long)T_ * H_ * D_;

  // ---- stage Q tile [64 x 128] -> LDS bf16 ----
  const float* qbase = q + (long)b * bstride + (long)q0 * H_ * D_ + h * D_;
  for (int e = tid; e < 64 * 32; e += 128) {
    const int row = e >> 5, d4 = (e & 31) * 4;
    const float4 qv = *(const float4*)(qbase + (long)row * H_ * D_ + d4);
    *(unsigned int*)&sQ[row * LDQ + d4]     = pk2bf(qv.x, qv.y);
    *(unsigned int*)&sQ[row * LDQ + d4 + 2] = pk2bf(qv.z, qv.w);
  }
  __syncthreads();

  // ---- build Q A-fragments (16x32 bf16 per 32-wide d-slice) ----
  v16u qa[4];
  {
    const int qrow = wave * 16 + mrow;
    #pragma unroll
    for (int s = 0; s < 4; ++s)
      qa[s] = cat8(*(const v8u*)&sQ[qrow * LDQ + s * 32 + abase],
                   *(const v8u*)&sQ[qrow * LDQ + s * 32 + 16 + abase]);
  }

  // constant all-ones B fragment (bf16 1.0 splat) for row-sum WMMA
  v16u bones;
  #pragma unroll
  for (int i = 0; i < 16; ++i) bones[i] = 0x3F80;

  const v8f vzero = {0.f, 0.f, 0.f, 0.f, 0.f, 0.f, 0.f, 0.f};
  v8f O[8];
  #pragma unroll
  for (int i = 0; i < 8; ++i) O[i] = vzero;
  float mst[8], lst[8];
  #pragma unroll
  for (int r = 0; r < 8; ++r) { mst[r] = -1e30f; lst[r] = 0.0f; }

  const int idx = *cur_idx;
  const int rk  = *run_kv;
  int start = idx - MAXC_; if (start < 0) start = 0;
  const int ncached = (rk != 0 && idx > 0) ? (idx - start) : 0;
  const int ntiles  = (ncached + 1) * TPC;
  // scores scaled into log2 domain: S * (1/sqrt(D)) * log2(e)
  const float scale2 = 0.08838834764831845f * 1.4426950408889634f;

  auto kchunk = [&](int t) -> const float* {
    const int c = t / TPC;
    return (c < ncached) ? (kc + ((long)(start + c) * B_ + b) * bstride)
                         : (kin + (long)b * bstride);
  };
  auto vchunk = [&](int t) -> const float* {
    const int c = t / TPC;
    return (c < ncached) ? (vc + ((long)(start + c) * B_ + b) * bstride)
                         : (vin + (long)b * bstride);
  };

  unsigned int kpk[8][2], vpk[8][2];
  auto load_tile = [&](int t) {
    const float* kb = kchunk(t);
    const float* vb = vchunk(t);
    const int kt = t % TPC;
    #pragma unroll
    for (int i = 0; i < 8; ++i) {
      const int e = tid + i * 128;
      const int row = e >> 5, d4 = (e & 31) * 4;
      const long goff = (long)(kt * KT + row) * H_ * D_ + h * D_ + d4;
      const float4 kv = *(const float4*)(kb + goff);
      kpk[i][0] = pk2bf(kv.x, kv.y);
      kpk[i][1] = pk2bf(kv.z, kv.w);
      const float4 vv = *(const float4*)(vb + goff);
      vpk[i][0] = pk2bf(vv.x, vv.y);
      vpk[i][1] = pk2bf(vv.z, vv.w);
    }
  };

  load_tile(0);  // prologue

  for (int t = 0; t < ntiles; ++t) {
    const int buf = t & 1;
    // ---- store staged regs -> LDS (K row-major bf16, V transposed) ----
    #pragma unroll
    for (int i = 0; i < 8; ++i) {
      const int e = tid + i * 128;
      const int row = e >> 5, d4 = (e & 31) * 4;
      *(unsigned int*)&sK[buf][row * LDK + d4]     = kpk[i][0];
      *(unsigned int*)&sK[buf][row * LDK + d4 + 2] = kpk[i][1];
      sVt[buf][(d4 + 0) * LDV + row] = (unsigned short)(vpk[i][0]);
      sVt[buf][(d4 + 1) * LDV + row] = (unsigned short)(vpk[i][0] >> 16);
      sVt[buf][(d4 + 2) * LDV + row] = (unsigned short)(vpk[i][1]);
      sVt[buf][(d4 + 3) * LDV + row] = (unsigned short)(vpk[i][1] >> 16);
    }
    __syncthreads();

    // ---- overlap: pull next tile from HBM while computing this one ----
    if (t + 1 < ntiles) load_tile(t + 1);
    if (t + 2 < ntiles) {  // global_prefetch_b8 for t+2
      const float* pk_ = kchunk(t + 2);
      const float* pv_ = vchunk(t + 2);
      const long poff = (long)(((t + 2) % TPC) * KT + (tid >> 2)) * H_ * D_ + h * D_ + (tid & 3) * 32;
      __builtin_prefetch(pk_ + poff, 0, 1);
      __builtin_prefetch(pv_ + poff, 0, 1);
    }

    // ---- S = (Q . K^T) in log2 units; two 16-key halves ----
    v8f S0 = vzero, S1 = vzero;
    #pragma unroll
    for (int sp = 0; sp < 2; ++sp) {   // 8 ds_loads -> fence -> 4 wmmas
      v16u bk[4];
      #pragma unroll
      for (int j = 0; j < 2; ++j) {
        const int s = sp * 2 + j;
        const int koff = s * 32 + hi16 * 8;  // B: lanes<16 -> K 0..7,16..23, lanes>=16 -> 8..15,24..31
        bk[2*j]   = cat8(*(const v8u*)&sK[buf][mrow * LDK + koff],
                         *(const v8u*)&sK[buf][mrow * LDK + koff + 16]);
        bk[2*j+1] = cat8(*(const v8u*)&sK[buf][(16 + mrow) * LDK + koff],
                         *(const v8u*)&sK[buf][(16 + mrow) * LDK + koff + 16]);
      }
      SCHED_FENCE();
      #pragma unroll
      for (int j = 0; j < 2; ++j) {
        const int s = sp * 2 + j;
        S0 = __builtin_amdgcn_wmma_f32_16x16x32_bf16(
                 false, __builtin_bit_cast(v16bf, qa[s]),
                 false, __builtin_bit_cast(v16bf, bk[2*j]), (short)0, S0, false, false);
        S1 = __builtin_amdgcn_wmma_f32_16x16x32_bf16(
                 false, __builtin_bit_cast(v16bf, qa[s]),
                 false, __builtin_bit_cast(v16bf, bk[2*j+1]), (short)0, S1, false, false);
      }
      SCHED_FENCE();
    }

    // ---- online softmax (exp2 domain), per C-layout row r + 8*hi16 ----
    float cr[8];  // per-row rescale factors, consumed after the row-sum WMMA
    #pragma unroll
    for (int r = 0; r < 8; ++r) {
      float s0 = S0[r] * scale2, s1 = S1[r] * scale2;
      const float vmax = red_max16(fmaxf(s0, s1));
      const float mn = fmaxf(mst[r], vmax);
      const float corr = exp2f(mst[r] - mn);
      mst[r] = mn;
      cr[r] = corr;
      const float p0 = exp2f(s0 - mn);
      const float p1 = exp2f(s1 - mn);
      const int prow = wave * 16 + r + hi16 * 8;   // wave-private P scratch
      sP[prow * LDP + mrow]      = f2bf(p0);
      sP[prow * LDP + 16 + mrow] = f2bf(p1);
      #pragma unroll
      for (int dt = 0; dt < 8; ++dt) O[dt][r] *= corr;
    }
    // P is wave-private: same-wave DS ordering + dscnt wait is sufficient, no barrier.

    // ---- O += P . V ; row-sum(P) via P x ones on the matrix pipe ----
    v16u pa;
    {
      const int prow = wave * 16 + mrow;
      pa = cat8(*(const v8u*)&sP[prow * LDP + abase],
                *(const v8u*)&sP[prow * LDP + 16 + abase]);
    }
    // Ssum tile: every column holds rowsum(P) -> each lane's reg r is its row's sum
    v8f Ssum = __builtin_amdgcn_wmma_f32_16x16x32_bf16(
                   false, __builtin_bit_cast(v16bf, pa),
                   false, __builtin_bit_cast(v16bf, bones), (short)0, vzero, false, false);
    #pragma unroll
    for (int g = 0; g < 2; ++g) {      // 8 ds_loads -> fence -> 4 wmmas
      v16u bv[4];
      #pragma unroll
      for (int j = 0; j < 4; ++j) {
        const int dcol = (g * 4 + j) * 16 + mrow;
        bv[j] = cat8(*(const v8u*)&sVt[buf][dcol * LDV + hi16 * 8],
                     *(const v8u*)&sVt[buf][dcol * LDV + 16 + hi16 * 8]);
      }
      SCHED_FENCE();
      #pragma unroll
      for (int j = 0; j < 4; ++j) {
        const int dt = g * 4 + j;
        O[dt] = __builtin_amdgcn_wmma_f32_16x16x32_bf16(
                    false, __builtin_bit_cast(v16bf, pa),
                    false, __builtin_bit_cast(v16bf, bv[j]), (short)0, O[dt], false, false);
      }
      SCHED_FENCE();
    }
    #pragma unroll
    for (int r = 0; r < 8; ++r) lst[r] = lst[r] * cr[r] + Ssum[r];
  }

  // ---- normalize and write output [B, T, H*D] ----
  #pragma unroll
  for (int r = 0; r < 8; ++r) lst[r] = 1.0f / lst[r];
  float* ob = out + ((long)b * T_ + q0 + wave * 16) * (H_ * D_) + h * D_;
  #pragma unroll
  for (int dt = 0; dt < 8; ++dt) {
    const int dcol = dt * 16 + mrow;
    #pragma unroll
    for (int r = 0; r < 8; ++r) {
      const int row = r + hi16 * 8;
      ob[(long)row * (H_ * D_) + dcol] = O[dt][r] * lst[r];
    }
  }
}

extern "C" void kernel_launch(void* const* d_in, const int* in_sizes, int n_in,
                              void* d_out, int out_size, void* d_ws, size_t ws_size,
                              hipStream_t stream) {
  const float* q  = (const float*)d_in[0];
  const float* k  = (const float*)d_in[1];
  const float* v  = (const float*)d_in[2];
  const float* kc = (const float*)d_in[3];
  const float* vc = (const float*)d_in[4];
  const int* cur  = (const int*)d_in[5];
  const int* sk   = (const int*)d_in[6];
  const int* rk   = (const int*)d_in[7];

  float* out = (float*)d_out;
  const size_t out0    = (size_t)B_ * T_ * H_ * D_;       // 4,194,304
  const size_t cachesz = (size_t)F_ * B_ * T_ * H_ * D_;  // 67,108,864
  float* out_kc = out + out0;
  float* out_vc = out_kc + cachesz;

  const unsigned nblk = (unsigned)(cachesz / 4 / 256);    // 65536
  cache_copy_kernel<<<dim3(nblk, 2), 256, 0, stream>>>(
      (const float4*)kc, (const float4*)vc, (const float4*)k, (const float4*)v,
      cur, sk, (float4*)out_kc, (float4*)out_vc);

  flash_attn_kernel<<<dim3(T_ / 64, H_, B_), 128, 0, stream>>>(
      q, k, v, kc, vc, cur, rk, out);
}
